// VectorQuantizer_10496900071525
// MI455X (gfx1250) — matrix operand order repaired and checked by reading, not compiled, not run on hardware
//
#include <hip/hip_runtime.h>
#include <hip/hip_bf16.h>

// ---------------------------------------------------------------------------
// VQ codebook lookup on gfx1250 via V_WMMA_F32_16X16X32_BF16.
//   d[l,n] = ||e_n||^2 - 2 * z_l . e_n      (||z_l||^2 dropped: row-constant)
//   idx[l] = argmin_n d[l,n] ; z_q = e[idx] ; loss = 1.25*mean((z_q - z)^2)
// Main loop processes TWO independent 16x16 column tiles per iteration so the
// scheduler can fill the 4-slot WMMA->VALU RAW hazard window of one tile with
// the other tile's loads / argmin VALU work.
// ---------------------------------------------------------------------------

typedef __attribute__((ext_vector_type(16))) __bf16 v16bf;
typedef __attribute__((ext_vector_type(8)))  float  v8f;

#define VQ_L 65536
#define VQ_D 64
#define VQ_N 1024

__device__ __forceinline__ unsigned short f2bf_rne(float x) {
    unsigned u = __float_as_uint(x);
    unsigned r = u + 0x7FFFu + ((u >> 16) & 1u);
    return (unsigned short)(r >> 16);
}

// --- Kernel 1: embedding -> bf16 bits + fp32 squared norms -------------------
__global__ void vq_prep_kernel(const float* __restrict__ emb,
                               unsigned short* __restrict__ ebf,
                               float* __restrict__ enorm) {
    int n = blockIdx.x * blockDim.x + threadIdx.x;
    if (n < VQ_N) {
        const float* r = emb + n * VQ_D;
        float s = 0.0f;
        #pragma unroll
        for (int k = 0; k < VQ_D; ++k) {
            float v = r[k];
            s += v * v;
            ebf[n * VQ_D + k] = f2bf_rne(v);
        }
        enorm[n] = s;
    }
}

// --- Kernel 2: WMMA distance + argmin + gather + loss partials ---------------
// 256 threads = 8 waves; wave w handles rows [bid*128 + w*16, +16).
__global__ __launch_bounds__(256)
void vq_main_kernel(const float* __restrict__ z,
                    const float* __restrict__ emb,          // fp32 codebook
                    const unsigned short* __restrict__ ebf, // bf16 codebook
                    const float* __restrict__ enorm,        // ||e_n||^2
                    float* __restrict__ out_zq,             // [L*D]
                    float* __restrict__ out_idx,            // [L] (as float)
                    float* __restrict__ partial)            // [gridDim.x]
{
    __shared__ int   sIdx[8][16];
    __shared__ float sPart[8];

    const int tid  = threadIdx.x;
    const int w    = tid >> 5;
    const int lane = tid & 31;
    const int half = lane >> 4;       // 0: lanes 0-15, 1: lanes 16-31
    const int l16  = lane & 15;
    const int row0 = blockIdx.x * 128 + w * 16;

    // ---- Load A fragments: z rows row0..row0+15, K=0..63 (bf16, contiguous K order)
    // lane<16: M=lane, elems 0..15 -> K 0..15 (+kOff)
    // lane>=16: M=lane-16, K base +16.
    union AF { v16bf v; unsigned short u[16]; } a0, a1;
    {
        const float* zr = z + (size_t)(row0 + l16) * VQ_D + half * 16;
        #pragma unroll
        for (int j = 0; j < 8; ++j) {
            a0.u[j]     = f2bf_rne(zr[j]);
            a0.u[8 + j] = f2bf_rne(zr[8 + j]);
            a1.u[j]     = f2bf_rne(zr[32 + j]);
            a1.u[8 + j] = f2bf_rne(zr[40 + j]);
        }
    }

    // ---- Loop over 64 column tiles of 16 codes (2 per iteration);
    // track per-slot running min.
    float mv[8];
    int   mi[8];
    #pragma unroll
    for (int s = 0; s < 8; ++s) { mv[s] = 3.4e38f; mi[s] = 0; }

    const int kb2 = half * 16;  // B fragment: lane supplies K=kb2..kb2+15 of col n
    union BF { v16bf v; uint4 q[2]; };

    for (int t = 0; t < VQ_N / 16; t += 2) {
        const int nA = (t << 4) + l16;        // tile t   : this lane's N
        const int nB = nA + 16;               // tile t+1 : this lane's N

        BF a_b0, a_b1, b_b0, b_b1;
        {
            const uint4* pA0 = (const uint4*)(ebf + (size_t)nA * VQ_D + kb2);
            const uint4* pA1 = (const uint4*)(ebf + (size_t)nA * VQ_D + 32 + kb2);
            const uint4* pB0 = (const uint4*)(ebf + (size_t)nB * VQ_D + kb2);
            const uint4* pB1 = (const uint4*)(ebf + (size_t)nB * VQ_D + 32 + kb2);
            a_b0.q[0] = pA0[0]; a_b0.q[1] = pA0[1];
            a_b1.q[0] = pA1[0]; a_b1.q[1] = pA1[1];
            b_b0.q[0] = pB0[0]; b_b0.q[1] = pB0[1];
            b_b1.q[0] = pB1[0]; b_b1.q[1] = pB1[1];
        }
        const float enA = enorm[nA];
        const float enB = enorm[nB];

        v8f cA = {0.f, 0.f, 0.f, 0.f, 0.f, 0.f, 0.f, 0.f};
        v8f cB = {0.f, 0.f, 0.f, 0.f, 0.f, 0.f, 0.f, 0.f};
        cA = __builtin_amdgcn_wmma_f32_16x16x32_bf16(false, a0.v, false, a_b0.v,
                                                     (short)0, cA, false, false);
        cB = __builtin_amdgcn_wmma_f32_16x16x32_bf16(false, a0.v, false, b_b0.v,
                                                     (short)0, cB, false, false);
        cA = __builtin_amdgcn_wmma_f32_16x16x32_bf16(false, a1.v, false, a_b1.v,
                                                     (short)0, cA, false, false);
        cB = __builtin_amdgcn_wmma_f32_16x16x32_bf16(false, a1.v, false, b_b1.v,
                                                     (short)0, cB, false, false);

        #pragma unroll
        for (int s = 0; s < 8; ++s) {
            float dA = fmaf(-2.0f, cA[s], enA);
            if (dA < mv[s]) { mv[s] = dA; mi[s] = nA; }
        }
        #pragma unroll
        for (int s = 0; s < 8; ++s) {
            float dB = fmaf(-2.0f, cB[s], enB);
            if (dB < mv[s]) { mv[s] = dB; mi[s] = nB; }
        }
    }

    // ---- Cross-lane argmin within each 16-lane half (C slot s: lanes 0-15
    // hold M=s, lanes 16-31 hold M=s+8). Tie-break: smaller index (argmin
    // first-occurrence semantics).
    #pragma unroll
    for (int s = 0; s < 8; ++s) {
        float v = mv[s];
        int   i = mi[s];
        #pragma unroll
        for (int m = 8; m >= 1; m >>= 1) {
            float ov = __shfl_xor(v, m, 32);
            int   oi = __shfl_xor(i, m, 32);
            if (ov < v || (ov == v && oi < i)) { v = ov; i = oi; }
        }
        if (lane == 0)  sIdx[w][s]     = i;   // row row0 + s
        if (lane == 16) sIdx[w][8 + s] = i;   // row row0 + 8 + s
    }
    __syncthreads();

    // ---- Indices output (as float), gather z_q from fp32 codebook, loss acc.
    if (lane < 16) out_idx[row0 + lane] = (float)sIdx[w][lane];

    float acc = 0.0f;
    const int col = lane * 2;   // each lane covers 2 of the 64 columns
    for (int r = 0; r < 16; ++r) {
        const int row = row0 + r;
        const int ci  = sIdx[w][r];
        float2 e2 = *(const float2*)(emb + (size_t)ci * VQ_D + col);
        float2 zz = *(const float2*)(z + (size_t)row * VQ_D + col);
        float dx = e2.x - zz.x, dy = e2.y - zz.y;
        acc = fmaf(dx, dx, acc);
        acc = fmaf(dy, dy, acc);
        *(float2*)(out_zq + (size_t)row * VQ_D + col) = e2;
    }

    // ---- Reduce loss partial: wave, then block.
    #pragma unroll
    for (int m = 16; m >= 1; m >>= 1) acc += __shfl_xor(acc, m, 32);
    if (lane == 0) sPart[w] = acc;
    __syncthreads();
    if (tid == 0) {
        float s = 0.0f;
        #pragma unroll
        for (int i = 0; i < 8; ++i) s += sPart[i];
        partial[blockIdx.x] = s;
    }
}

// --- Kernel 3: final loss reduction ------------------------------------------
__global__ void vq_loss_kernel(const float* __restrict__ partial,
                               float* __restrict__ out_loss) {
    __shared__ float sm[512];
    int t = threadIdx.x;
    sm[t] = partial[t];
    __syncthreads();
    for (int s = 256; s > 0; s >>= 1) {
        if (t < s) sm[t] += sm[t + s];
        __syncthreads();
    }
    if (t == 0) {
        // loss = (beta + 1) * mean((z_q - z)^2) = 1.25 * sum / (L*D)
        out_loss[0] = sm[0] * (1.25f / (float)(VQ_L * VQ_D));
    }
}

extern "C" void kernel_launch(void* const* d_in, const int* in_sizes, int n_in,
                              void* d_out, int out_size, void* d_ws, size_t ws_size,
                              hipStream_t stream) {
    const float* z   = (const float*)d_in[0];   // [65536, 64] fp32
    const float* emb = (const float*)d_in[1];   // [1024, 64] fp32

    float* out = (float*)d_out;
    float* out_zq   = out;                       // [L*D]
    float* out_loss = out + (size_t)VQ_L * VQ_D; // [1]
    float* out_idx  = out_loss + 1;              // [L]

    // Workspace layout.
    unsigned short* ebf   = (unsigned short*)d_ws;                 // 131072 B
    float*          enorm = (float*)((char*)d_ws + 131072);        //   4096 B
    float*          part  = (float*)((char*)d_ws + 131072 + 4096); //   2048 B

    vq_prep_kernel<<<(VQ_N + 255) / 256, 256, 0, stream>>>(emb, ebf, enorm);
    vq_main_kernel<<<VQ_L / 128, 256, 0, stream>>>(z, emb, ebf, enorm,
                                                   out_zq, out_idx, part);
    vq_loss_kernel<<<1, 512, 0, stream>>>(part, out_loss);
}
